// LinearAttention_38749194944572
// MI455X (gfx1250) — matrix operand order repaired and checked by reading, not compiled, not run on hardware
//
#include <hip/hip_runtime.h>

typedef __attribute__((ext_vector_type(16))) __bf16 v16bf;
typedef __attribute__((ext_vector_type(8)))  float  v8f;
typedef __attribute__((ext_vector_type(4)))  unsigned int v4u;
typedef __attribute__((ext_vector_type(8)))  unsigned int v8u;

#define B_   8
#define N_   4096
#define DIM_ 512
#define H_   8
#define DH_  64

static __device__ __forceinline__ unsigned short f2bf(float f) {
  unsigned int u = __float_as_uint(f);
  u = u + 0x7FFFu + ((u >> 16) & 1u);      // round-to-nearest-even
  return (unsigned short)(u >> 16);
}
static __device__ __forceinline__ __bf16 us2bf(unsigned short u) {
  union { unsigned short u; __bf16 b; } x; x.u = u; return x.b;
}

// ---------------------------------------------------------------------------
// Tensor Data Mover: issue a 2-D (or 1-D when td1==1) tile load of 2-byte
// elements from global memory into LDS. Descriptor per CDNA5 ISA §8.3/8.4:
//   group0: [count=1 | lds_addr | global_addr(57b) | type=2]
//   group1: [wg_mask=0, data_size=1(2B) | tensor dims = tile dims (no OOB
//            clip) | tile dims | dim0 stride (elements)]
// 2-group form (VADDR2/VADDR3 = NULL) is valid for tensors up to 2-D.
// Tracked by TENSORcnt; caller waits with s_wait_tensorcnt before barrier.
// ---------------------------------------------------------------------------
static __device__ __forceinline__ void tdm_load_bf16(
    unsigned lds_addr, const unsigned short* gptr,
    unsigned td0, unsigned td1, unsigned stride0_elems)
{
  unsigned long long ga = (unsigned long long)(size_t)gptr;
  v4u g0;
  g0.x = 1u;                                                  // count=1
  g0.y = lds_addr;                                            // LDS byte addr
  g0.z = (unsigned)ga;                                        // gaddr[31:0]
  g0.w = (unsigned)((ga >> 32) & 0x01FFFFFFu) | (2u << 30);   // gaddr[56:32]|type=2
  v8u g1;
  g1[0] = 1u << 16;                                           // data_size=2B
  g1[1] = (td0 & 0xFFFFu) << 16;                              // tensor_dim0 lo
  g1[2] = ((td0 >> 16) & 0xFFFFu) | ((td1 & 0xFFFFu) << 16);  // td0 hi|td1 lo
  g1[3] = ((td1 >> 16) & 0xFFFFu) | ((td0 & 0xFFFFu) << 16);  // td1 hi|tile_dim0
  g1[4] = td1 & 0xFFFFu;                                      // tile_dim1, tile_dim2=0
  g1[5] = stride0_elems;                                      // dim0 stride lo32
  g1[6] = 0u;                                                 // stride hi16, dim1 stride lo
  g1[7] = 0u;                                                 // dim1 stride hi
  asm volatile("tensor_load_to_lds %0, %1" :: "s"(g0), "s"(g1) : "memory");
}

// ---------------------------------------------------------------------------
// Kernel 1: qkv = x @ W_qkv^T, then per-head LayerNorm (k,v) + 2D RoPE (q,k),
// store q/k/v as bf16 in (B,H,N,DH) layout.
// Grid: (B*N/64, 24)  Block: 128 (4 waves). Tile: 64 tokens x 64 features
// (one head-chunk), K loop over DIM in steps of 32 (bf16 WMMA 16x16x32).
// ---------------------------------------------------------------------------
__global__ __launch_bounds__(128) void k_qkv_ln_rope(
    const float* __restrict__ x, const float* __restrict__ pos,
    const float* __restrict__ Wqkv,
    const float* __restrict__ gk, const float* __restrict__ bk,
    const float* __restrict__ gv, const float* __restrict__ bv,
    unsigned short* __restrict__ qo, unsigned short* __restrict__ ko,
    unsigned short* __restrict__ vo)
{
  __shared__ __align__(16) __bf16 Xs[64][32];
  __shared__ __align__(16) __bf16 Ws[64][32];
  __shared__ float  Ts[64][65];

  const int tid  = threadIdx.x;
  const int lane = tid & 31, wv = tid >> 5;
  const int g = lane >> 4, mr = lane & 15;

  const int tileM = blockIdx.x;          // token tile
  const int chunk = blockIdx.y;          // 0..23 (q:0-7, k:8-15, v:16-23)
  const int type  = chunk >> 3, h = chunk & 7;
  const int tok0  = tileM * 64;
  const int b     = tok0 / N_, n0 = tok0 % N_;
  const int c0    = chunk * 64;          // feature column base in qkv

  v8f acc[4] = {};

  for (int k0 = 0; k0 < DIM_; k0 += 32) {
    for (int i = tid; i < 64 * 32; i += 128) {
      int r = i >> 5, c = i & 31;
      Xs[r][c] = us2bf(f2bf(x[((size_t)(b * N_ + n0 + r)) * DIM_ + k0 + c]));
      Ws[r][c] = us2bf(f2bf(Wqkv[(size_t)(c0 + r) * DIM_ + k0 + c]));
    }
    __syncthreads();

    v16bf a;
#pragma unroll
    for (int e = 0; e < 16; ++e) {
      int kk = (e < 8) ? (g * 8 + e) : (g * 8 + e + 8);
      a[e] = Xs[wv * 16 + mr][kk];
    }
#pragma unroll
    for (int nt = 0; nt < 4; ++nt) {
      v16bf bb;
#pragma unroll
      for (int e = 0; e < 16; ++e)
        bb[e] = Ws[nt * 16 + mr][(e < 8) ? (g * 8 + e) : (g * 8 + e + 8)];
      acc[nt] = __builtin_amdgcn_wmma_f32_16x16x32_bf16(
          false, a, false, bb, (short)0, acc[nt], false, false);
    }
    __syncthreads();
  }

  // spill accumulators: row = wv*16 + g*8 + r, col = nt*16 + mr
#pragma unroll
  for (int nt = 0; nt < 4; ++nt)
#pragma unroll
    for (int r = 0; r < 8; ++r)
      Ts[wv * 16 + g * 8 + r][nt * 16 + mr] = acc[nt][r];
  __syncthreads();

  // epilogue: 2 threads per row, 32 columns each
  const int row = tid >> 1, hf = tid & 1;
  float vals[32];
#pragma unroll
  for (int j = 0; j < 32; ++j) vals[j] = Ts[row][hf * 32 + j];

  if (type >= 1) {  // LayerNorm for k and v (over DH=64, population variance)
    float s = 0.f, sq = 0.f;
#pragma unroll
    for (int j = 0; j < 32; ++j) { s += vals[j]; sq += vals[j] * vals[j]; }
    s  += __shfl_xor(s, 1);
    sq += __shfl_xor(sq, 1);
    float mu  = s * (1.f / 64.f);
    float var = sq * (1.f / 64.f) - mu * mu;
    float inv = rsqrtf(var + 1e-5f);
    const float* gam = (type == 1) ? gk : gv;
    const float* bet = (type == 1) ? bk : bv;
#pragma unroll
    for (int j = 0; j < 32; ++j)
      vals[j] = (vals[j] - mu) * inv * gam[hf * 32 + j] + bet[hf * 32 + j];
  }

  if (type <= 1) {  // 2D RoPE for q and k: half 0 uses pos.x, half 1 uses pos.y
    float coord = pos[((size_t)(b * N_ + n0 + row)) * 2 + hf];
    float t = coord * 64.f;  // SCALE / MIN_FREQ
#pragma unroll
    for (int i = 0; i < 16; ++i) {
      float fr = t * __powf(10000.f, -(float)i * (1.f / 16.f));
      float sn, cs;
      __sincosf(fr, &sn, &cs);
      float lo = vals[i], hi = vals[i + 16];
      vals[i]      = lo * cs - hi * sn;
      vals[i + 16] = hi * cs + lo * sn;
    }
  }

  unsigned short* dst = (type == 0) ? qo : (type == 1) ? ko : vo;
  size_t base = (((size_t)(b * H_ + h)) * N_ + n0 + row) * DH_ + hf * 32;
#pragma unroll
  for (int j = 0; j < 32; ++j) dst[base + j] = f2bf(vals[j]);
}

// ---------------------------------------------------------------------------
// Kernel 2: dots[b,h] = K^T @ V  (64x64 per head), reduce over N=4096.
// K/V tiles staged via TDM (contiguous 4KB bf16 tiles). 1/N folded into the
// bf16 conversion. Grid: (B*H), Block: 128.
// ---------------------------------------------------------------------------
__global__ __launch_bounds__(128) void k_kv_dots(
    const unsigned short* __restrict__ kin, const unsigned short* __restrict__ vin,
    unsigned short* __restrict__ dots)
{
  __shared__ __align__(16) __bf16 Ks[32][64];  // [n-slice][d]
  __shared__ __align__(16) __bf16 Vs[32][64];  // [n-slice][e]

  const int tid  = threadIdx.x;
  const int lane = tid & 31, wv = tid >> 5;
  const int g = lane >> 4, mr = lane & 15;
  const int bh = blockIdx.x;

  const unsigned short* kp = kin + (size_t)bh * N_ * DH_;
  const unsigned short* vp = vin + (size_t)bh * N_ * DH_;
  const unsigned ldsK = (unsigned)(size_t)&Ks[0][0];
  const unsigned ldsV = (unsigned)(size_t)&Vs[0][0];

  v8f acc[4] = {};
  for (int n0 = 0; n0 < N_; n0 += 32) {
    if (wv == 0) {  // one wave drives the DMA; TDM ignores EXEC
      tdm_load_bf16(ldsK, kp + (size_t)n0 * DH_, 32 * DH_, 1, 32 * DH_);
      tdm_load_bf16(ldsV, vp + (size_t)n0 * DH_, 32 * DH_, 1, 32 * DH_);
      __builtin_amdgcn_s_wait_tensorcnt(0);
    }
    __syncthreads();

    v16bf a;  // A = K^T: A[m=d, kk=n] = K[n][d]
#pragma unroll
    for (int e = 0; e < 16; ++e) {
      int kk = (e < 8) ? (g * 8 + e) : (g * 8 + e + 8);
      a[e] = Ks[kk][wv * 16 + mr];
    }
#pragma unroll
    for (int nt = 0; nt < 4; ++nt) {
      v16bf bb;  // B = V: B[kk=n, nn=e]
#pragma unroll
      for (int e = 0; e < 16; ++e)
        bb[e] = Vs[(e < 8) ? (g * 8 + e) : (g * 8 + e + 8)][nt * 16 + mr];
      acc[nt] = __builtin_amdgcn_wmma_f32_16x16x32_bf16(
          false, a, false, bb, (short)0, acc[nt], false, false);
    }
    __syncthreads();
  }

  const float invN = 1.f / (float)N_;
#pragma unroll
  for (int nt = 0; nt < 4; ++nt)
#pragma unroll
    for (int r = 0; r < 8; ++r)
      dots[(size_t)bh * DH_ * DH_ + (wv * 16 + g * 8 + r) * DH_ + nt * 16 + mr] =
          f2bf(acc[nt][r] * invN);
}

// ---------------------------------------------------------------------------
// Kernel 3: out_h = Q @ dots, write to ctx in (B, N, H*DH) layout (bf16).
// Q and dots tiles staged via TDM (contiguous 8KB each).
// Grid: (B*H, N/64), Block: 128. K=64 -> two WMMA k-steps.
// ---------------------------------------------------------------------------
__global__ __launch_bounds__(128) void k_q_dots(
    const unsigned short* __restrict__ qin, const unsigned short* __restrict__ dots,
    unsigned short* __restrict__ ctx)
{
  __shared__ __align__(16) __bf16 Qs[64][64];
  __shared__ __align__(16) __bf16 Ds[64][64];

  const int tid  = threadIdx.x;
  const int lane = tid & 31, wv = tid >> 5;
  const int g = lane >> 4, mr = lane & 15;
  const int bh = blockIdx.x, b = bh >> 3, h = bh & 7;
  const int n0 = blockIdx.y * 64;

  const unsigned short* qp = qin  + (size_t)bh * N_ * DH_;
  const unsigned short* dp = dots + (size_t)bh * DH_ * DH_;

  if (wv == 0) {
    tdm_load_bf16((unsigned)(size_t)&Qs[0][0], qp + (size_t)n0 * DH_,
                  64 * DH_, 1, 64 * DH_);
    tdm_load_bf16((unsigned)(size_t)&Ds[0][0], dp, 64 * DH_, 1, 64 * DH_);
    __builtin_amdgcn_s_wait_tensorcnt(0);
  }
  __syncthreads();

  v8f acc[4] = {};
#pragma unroll
  for (int ks = 0; ks < 64; ks += 32) {
    v16bf a;
#pragma unroll
    for (int e = 0; e < 16; ++e) {
      int kk = (e < 8) ? (g * 8 + e) : (g * 8 + e + 8);
      a[e] = Qs[wv * 16 + mr][ks + kk];
    }
#pragma unroll
    for (int nt = 0; nt < 4; ++nt) {
      v16bf bb;
#pragma unroll
      for (int e = 0; e < 16; ++e)
        bb[e] = Ds[ks + ((e < 8) ? (g * 8 + e) : (g * 8 + e + 8))][nt * 16 + mr];
      acc[nt] = __builtin_amdgcn_wmma_f32_16x16x32_bf16(
          false, a, false, bb, (short)0, acc[nt], false, false);
    }
  }

#pragma unroll
  for (int nt = 0; nt < 4; ++nt)
#pragma unroll
    for (int r = 0; r < 8; ++r) {
      int row = wv * 16 + g * 8 + r, col = nt * 16 + mr;
      ctx[((size_t)(b * N_ + n0 + row)) * (H_ * DH_) + h * DH_ + col] =
          f2bf(acc[nt][r]);
    }
}

// ---------------------------------------------------------------------------
// Kernel 4: out = ctx @ W_out^T + b_out  (f32 output).
// ctx tile staged via 2-D TDM (64 rows x 32 cols, stride 512 elements);
// W_out tile needs f32->bf16 conversion so stays on the VALU fill path.
// Grid: (B*N/64, DIM/64), Block: 128.
// ---------------------------------------------------------------------------
__global__ __launch_bounds__(128) void k_out_proj(
    const unsigned short* __restrict__ ctx, const float* __restrict__ Wout,
    const float* __restrict__ bout, float* __restrict__ out)
{
  __shared__ __align__(16) __bf16 Cs[64][32];
  __shared__ __align__(16) __bf16 Ws[64][32];

  const int tid  = threadIdx.x;
  const int lane = tid & 31, wv = tid >> 5;
  const int g = lane >> 4, mr = lane & 15;
  const int t0 = blockIdx.x * 64;
  const int o0 = blockIdx.y * 64;
  const unsigned ldsC = (unsigned)(size_t)&Cs[0][0];

  v8f acc[4] = {};
  for (int k0 = 0; k0 < DIM_; k0 += 32) {
    if (wv == 0)  // 2-D tile: 32 elems/row, 64 rows, row stride = DIM_
      tdm_load_bf16(ldsC, ctx + (size_t)t0 * DIM_ + k0, 32, 64, DIM_);
    for (int i = tid; i < 64 * 32; i += 128) {
      int r = i >> 5, c = i & 31;
      Ws[r][c] = us2bf(f2bf(Wout[(size_t)(o0 + r) * DIM_ + k0 + c]));
    }
    if (wv == 0) __builtin_amdgcn_s_wait_tensorcnt(0);
    __syncthreads();

    v16bf a;
#pragma unroll
    for (int e = 0; e < 16; ++e) {
      int kk = (e < 8) ? (g * 8 + e) : (g * 8 + e + 8);
      a[e] = Cs[wv * 16 + mr][kk];
    }
#pragma unroll
    for (int nt = 0; nt < 4; ++nt) {
      v16bf bb;
#pragma unroll
      for (int e = 0; e < 16; ++e)
        bb[e] = Ws[nt * 16 + mr][(e < 8) ? (g * 8 + e) : (g * 8 + e + 8)];
      acc[nt] = __builtin_amdgcn_wmma_f32_16x16x32_bf16(
          false, a, false, bb, (short)0, acc[nt], false, false);
    }
    __syncthreads();
  }

#pragma unroll
  for (int nt = 0; nt < 4; ++nt)
#pragma unroll
    for (int r = 0; r < 8; ++r) {
      int row = wv * 16 + g * 8 + r, col = nt * 16 + mr;
      out[(size_t)(t0 + row) * DIM_ + o0 + col] = acc[nt][r] + bout[o0 + col];
    }
}

// ---------------------------------------------------------------------------
extern "C" void kernel_launch(void* const* d_in, const int* in_sizes, int n_in,
                              void* d_out, int out_size, void* d_ws, size_t ws_size,
                              hipStream_t stream)
{
  (void)in_sizes; (void)n_in; (void)out_size; (void)ws_size;

  const float* x    = (const float*)d_in[0];
  const float* pos  = (const float*)d_in[1];
  const float* Wqkv = (const float*)d_in[2];
  const float* gk   = (const float*)d_in[3];
  const float* bk   = (const float*)d_in[4];
  const float* gv   = (const float*)d_in[5];
  const float* bv   = (const float*)d_in[6];
  const float* Wout = (const float*)d_in[7];
  const float* bout = (const float*)d_in[8];
  float* out = (float*)d_out;

  const size_t QKV = (size_t)B_ * H_ * N_ * DH_;   // 16,777,216 elems
  const size_t DOT = (size_t)B_ * H_ * DH_ * DH_;  //    262,144 elems
  unsigned short* q    = (unsigned short*)d_ws;
  unsigned short* kbuf = q + QKV;
  unsigned short* vbuf = kbuf + QKV;
  unsigned short* dots = vbuf + QKV;
  unsigned short* ctx  = dots + DOT;

  dim3 blk(128);
  k_qkv_ln_rope<<<dim3((B_ * N_) / 64, 24), blk, 0, stream>>>(
      x, pos, Wqkv, gk, bk, gv, bv, q, kbuf, vbuf);
  k_kv_dots<<<dim3(B_ * H_), blk, 0, stream>>>(kbuf, vbuf, dots);
  k_q_dots<<<dim3(B_ * H_, N_ / 64), blk, 0, stream>>>(q, dots, ctx);
  k_out_proj<<<dim3((B_ * N_) / 64, DIM_ / 64), blk, 0, stream>>>(
      ctx, Wout, bout, out);
}
